// HopfieldNetwork_50568944943404
// MI455X (gfx1250) — 8-bit matrix operand order repaired and checked by reading, compiled, not run on hardware
//
#include <hip/hip_runtime.h>
#include <stdint.h>


typedef __attribute__((ext_vector_type(8))) int v8i;

#define HN_B    8192
#define HN_F    1024
#define HN_IT   30

__global__ void k_init_flag(int* flag) {
    flag[0] = 0;
    flag[1] = 0;
}

__global__ void k_cvt_w(const float* __restrict__ w, int8_t* __restrict__ w8) {
    int i = blockIdx.x * blockDim.x + threadIdx.x;
    if (i < HN_F * HN_F) w8[i] = (int8_t)__float2int_rn(w[i]);
}

__global__ void k_cvt_x(const float* __restrict__ x, int8_t* __restrict__ x8) {
    int i = blockIdx.x * blockDim.x + threadIdx.x;
    if (i < HN_B * HN_F) x8[i] = (int8_t)__float2int_rn(x[i]);
}

__global__ void k_zero_e(int* __restrict__ e, const int* __restrict__ flag) {
    if (flag[0]) return;
    int i = blockIdx.x * blockDim.x + threadIdx.x;
    if (i < HN_B) e[i] = 0;
}

__global__ __launch_bounds__(256)
void k_step(const int8_t* __restrict__ Xin, const int8_t* __restrict__ W8,
            int8_t* __restrict__ Xout, int* __restrict__ rowE,
            const int* __restrict__ flag)
{
    if (flag[0]) return;

    const int lane = threadIdx.x & 31;
    const int wave = threadIdx.x >> 5;
    const int l15  = lane & 15;
    const int half = lane >> 4;

    const int m0 = blockIdx.y * 128 + wave * 16;
    const int n0 = blockIdx.x * 64;

    const v8i vzero = {0, 0, 0, 0, 0, 0, 0, 0};
    v8i acc0 = vzero, acc1 = vzero, acc2 = vzero, acc3 = vzero;

    const int8_t* aptr = Xin + (size_t)(m0 + l15) * HN_F + 16 * half;
    const int8_t* bptr = W8 + (size_t)(n0 + l15) * HN_F + 16 * half;

#pragma unroll 1
    for (int k0 = 0; k0 < HN_F; k0 += 64) {
        int2 a01 = *reinterpret_cast<const int2*>(aptr + k0);
        int2 a23 = *reinterpret_cast<const int2*>(aptr + k0 + 8);
        int2 a45 = *reinterpret_cast<const int2*>(aptr + k0 + 32);
        int2 a67 = *reinterpret_cast<const int2*>(aptr + k0 + 40);
        v8i a;
        a[0] = a01.x; a[1] = a01.y; a[2] = a23.x; a[3] = a23.y;
        a[4] = a45.x; a[5] = a45.y; a[6] = a67.x; a[7] = a67.y;

        const int8_t* bp0 = bptr + k0;
        int4 b0lo = *reinterpret_cast<const int4*>(bp0);
        int4 b0hi = *reinterpret_cast<const int4*>(bp0 + 32);
        int4 b1lo = *reinterpret_cast<const int4*>(bp0 + 16 * HN_F);
        int4 b1hi = *reinterpret_cast<const int4*>(bp0 + 16 * HN_F + 32);
        int4 b2lo = *reinterpret_cast<const int4*>(bp0 + 32 * HN_F);
        int4 b2hi = *reinterpret_cast<const int4*>(bp0 + 32 * HN_F + 32);
        int4 b3lo = *reinterpret_cast<const int4*>(bp0 + 48 * HN_F);
        int4 b3hi = *reinterpret_cast<const int4*>(bp0 + 48 * HN_F + 32);

        v8i b0, b1, b2, b3;
        b0[0] = b0lo.x; b0[1] = b0lo.y; b0[2] = b0lo.z; b0[3] = b0lo.w;
        b0[4] = b0hi.x; b0[5] = b0hi.y; b0[6] = b0hi.z; b0[7] = b0hi.w;
        b1[0] = b1lo.x; b1[1] = b1lo.y; b1[2] = b1lo.z; b1[3] = b1lo.w;
        b1[4] = b1hi.x; b1[5] = b1hi.y; b1[6] = b1hi.z; b1[7] = b1hi.w;
        b2[0] = b2lo.x; b2[1] = b2lo.y; b2[2] = b2lo.z; b2[3] = b2lo.w;
        b2[4] = b2hi.x; b2[5] = b2hi.y; b2[6] = b2hi.z; b2[7] = b2hi.w;
        b3[0] = b3lo.x; b3[1] = b3lo.y; b3[2] = b3lo.z; b3[3] = b3lo.w;
        b3[4] = b3hi.x; b3[5] = b3hi.y; b3[6] = b3hi.z; b3[7] = b3hi.w;

        acc0 = __builtin_amdgcn_wmma_i32_16x16x64_iu8(true, a, true, b0, acc0,
                                                      false, false);
        acc1 = __builtin_amdgcn_wmma_i32_16x16x64_iu8(true, a, true, b1, acc1,
                                                      false, false);
        acc2 = __builtin_amdgcn_wmma_i32_16x16x64_iu8(true, a, true, b2, acc2,
                                                      false, false);
        acc3 = __builtin_amdgcn_wmma_i32_16x16x64_iu8(true, a, true, b3, acc3,
                                                      false, false);
    }

    const int mBase = m0 + 8 * half;
    int e[8];
#pragma unroll
    for (int r = 0; r < 8; ++r) e[r] = 0;

#pragma unroll
    for (int t = 0; t < 4; ++t) {
        const v8i acc = (t == 0) ? acc0 : (t == 1) ? acc1 : (t == 2) ? acc2
                                                                     : acc3;
        const int col = n0 + 16 * t + l15;
#pragma unroll
        for (int r = 0; r < 8; ++r) {
            const int y = acc[r];
            const size_t off = (size_t)(mBase + r) * HN_F + col;
            e[r] += (int)Xin[off] * y;
            Xout[off] = (int8_t)((y > 0) - (y < 0));
        }
    }

#pragma unroll
    for (int r = 0; r < 8; ++r) {
        int v = e[r];
        v += __shfl_xor(v, 1, 32);
        v += __shfl_xor(v, 2, 32);
        v += __shfl_xor(v, 4, 32);
        v += __shfl_xor(v, 8, 32);
        if (l15 == 0) atomicAdd(rowE + mBase + r, v);
    }
}

__global__ void k_check(const int* __restrict__ Ea, const int* __restrict__ Eb,
                        int* __restrict__ flag, int resultIdx)
{
    if (flag[0]) return;
    __shared__ int bad;
    if (threadIdx.x == 0) bad = 0;
    __syncthreads();
    int mism = 0;
    for (int i = threadIdx.x; i < HN_B; i += blockDim.x)
        mism |= (Ea[i] != Eb[i]);
    if (mism) atomicOr(&bad, 1);
    __syncthreads();
    if (threadIdx.x == 0 && bad == 0) {
        flag[0] = 1;
        flag[1] = resultIdx;
    }
}

__global__ void k_final(const int8_t* __restrict__ x0,
                        const int8_t* __restrict__ x1,
                        const int8_t* __restrict__ x2,
                        const int* __restrict__ flag,
                        float* __restrict__ out)
{
    const int idx = flag[0] ? flag[1] : 0;
    const int8_t* src = (idx == 0) ? x0 : (idx == 1) ? x1 : x2;
    int i = blockIdx.x * blockDim.x + threadIdx.x;
    if (i < HN_B * HN_F) out[i] = (float)src[i];
}

extern "C" void kernel_launch(void* const* d_in, const int* in_sizes, int n_in,
                              void* d_out, int out_size, void* d_ws, size_t ws_size,
                              hipStream_t stream)
{
    (void)in_sizes; (void)n_in; (void)out_size; (void)ws_size;
    const float* x = (const float*)d_in[0];
    const float* w = (const float*)d_in[1];

    uint8_t* ws = (uint8_t*)d_ws;
    const size_t XB = (size_t)HN_B * HN_F;
    int8_t* W8   = (int8_t*)(ws);
    int8_t* X[3] = { (int8_t*)(ws + (1u << 20)),
                     (int8_t*)(ws + (1u << 20) + XB),
                     (int8_t*)(ws + (1u << 20) + 2 * XB) };
    int* E[2];
    E[0]      = (int*)(ws + (1u << 20) + 3 * XB);
    E[1]      = E[0] + HN_B;
    int* flag = E[1] + HN_B;

    k_init_flag<<<1, 1, 0, stream>>>(flag);
    k_cvt_w<<<(HN_F * HN_F) / 256, 256, 0, stream>>>(w, W8);
    k_cvt_x<<<(HN_B * HN_F) / 256, 256, 0, stream>>>(x, X[0]);

    dim3 grid(HN_F / 64, HN_B / 128);

    k_zero_e<<<HN_B / 256, 256, 0, stream>>>(E[0], flag);
    k_step<<<grid, 256, 0, stream>>>(X[0], W8, X[1], E[0], flag);

    for (int i = 1; i <= HN_IT; ++i) {
        k_zero_e<<<HN_B / 256, 256, 0, stream>>>(E[i & 1], flag);
        k_step<<<grid, 256, 0, stream>>>(X[i % 3], W8, X[(i + 1) % 3],
                                         E[i & 1], flag);
        k_check<<<1, 1024, 0, stream>>>(E[(i - 1) & 1], E[i & 1], flag,
                                        (i - 1) % 3);
    }

    k_final<<<(HN_B * HN_F) / 256, 256, 0, stream>>>(X[0], X[1], X[2], flag,
                                                     (float*)d_out);
}
